// SimpleGCN_61229053772023
// MI455X (gfx1250) — matrix operand order repaired and checked by reading, not compiled, not run on hardware
//
#include <hip/hip_runtime.h>
#include <hip/hip_bf16.h>

// ---------------------------------------------------------------------------
// Types for CDNA5 WMMA (gfx1250, wave32)
// ---------------------------------------------------------------------------
typedef __attribute__((ext_vector_type(16))) __bf16 v16bf;
typedef __attribute__((ext_vector_type(8)))  __bf16 v8bf;
typedef __attribute__((ext_vector_type(8)))  float  v8f;

// async global->LDS builtin operand types (match clang's "vV4i*1V4i*3IiIi")
typedef int b128_t __attribute__((vector_size(16)));
typedef __attribute__((address_space(1))) b128_t* gptr128;
typedef __attribute__((address_space(3))) b128_t* lptr128;

#define IN_CH   128
#define HID     128
#define EMB     64

__device__ __forceinline__ __bf16 f2bf(float f) {
    // round-to-nearest-even float -> bf16
    unsigned int u = __builtin_bit_cast(unsigned int, f);
    u += 0x7FFFu + ((u >> 16) & 1u);
    unsigned short s = (unsigned short)(u >> 16);
    return __builtin_bit_cast(__bf16, s);
}

// ---------------------------------------------------------------------------
// Degree / normalization
// ---------------------------------------------------------------------------
__global__ void k_init_one(float* deg, int n) {
    int i = blockIdx.x * blockDim.x + threadIdx.x;
    if (i < n) deg[i] = 1.0f;  // self-loop contributes 1 to every node's degree
}

__global__ void k_degree(const int* __restrict__ ei, float* deg, int E) {
    int e = blockIdx.x * blockDim.x + threadIdx.x;
    if (e < E) atomicAdd(&deg[ei[E + e]], 1.0f);  // dst row of edge_index
}

__global__ void k_rsqrt(float* dinv, int n) {
    int i = blockIdx.x * blockDim.x + threadIdx.x;
    if (i < n) {
        float d = dinv[i];
        dinv[i] = (d > 0.0f) ? rsqrtf(d) : 0.0f;
    }
}

// ---------------------------------------------------------------------------
// fp32 -> bf16 conversions
// ---------------------------------------------------------------------------
__global__ void k_f2bf(const float* __restrict__ in, __bf16* __restrict__ out, int n) {
    int i = blockIdx.x * blockDim.x + threadIdx.x;
    if (i < n) out[i] = f2bf(in[i]);
}

// W is K x NOUT row-major fp32; produce Wt = NOUT x K row-major bf16 (K contiguous)
__global__ void k_transpose_bf(const float* __restrict__ W, __bf16* __restrict__ Wt,
                               int K, int NOUT) {
    int id = blockIdx.x * blockDim.x + threadIdx.x;
    if (id < K * NOUT) {
        int k = id / NOUT;
        int n = id % NOUT;
        Wt[n * K + k] = f2bf(W[k * NOUT + n]);
    }
}

// ---------------------------------------------------------------------------
// GEMM: C[nrows x NOUT] = A[nrows x 128](bf16) @ W[128 x NOUT]
// W supplied pre-transposed as Bt[NOUT x 128] bf16, staged into LDS (padded
// rows, async global->LDS). Block = 256 threads = 8 waves; wave w computes
// rows [blk*128 + w*16, +16) for all NOUT/16 column tiles, K = 128 in 4
// chunks of 32 via v_wmma_f32_16x16x32_bf16.
// ---------------------------------------------------------------------------
template <int NOUT>
__global__ void __launch_bounds__(256)
k_gemm_bf16_wmma(const __bf16* __restrict__ A, const __bf16* __restrict__ Bt,
                 float* __restrict__ C, int nrows) {
    constexpr int SBS = 136;              // padded LDS row stride (elements)
    __shared__ __bf16 sB[NOUT * SBS];

    // ---- stage Bt (NOUT x 128 bf16) into LDS, one b128 chunk per lane ----
    {
        const int nchunk = NOUT * 16;     // 16-byte chunks (8 bf16 each)
        for (int idx = threadIdx.x; idx < nchunk; idx += 256) {
            const int row = idx >> 4, ch = idx & 15;
            const __bf16* src = Bt + row * 128 + ch * 8;
            __bf16*       dst = sB + row * SBS + ch * 8;
#if __has_builtin(__builtin_amdgcn_global_load_async_to_lds_b128)
            __builtin_amdgcn_global_load_async_to_lds_b128(
                (gptr128)src, (lptr128)dst, 0, 0);
#else
            *(v8bf*)dst = *(const v8bf*)src;
#endif
        }
#if __has_builtin(__builtin_amdgcn_global_load_async_to_lds_b128)
#if __has_builtin(__builtin_amdgcn_s_wait_asynccnt)
        __builtin_amdgcn_s_wait_asynccnt(0);
#else
        asm volatile("s_wait_asynccnt 0x0" ::: "memory");
#endif
#endif
        __syncthreads();
    }

    const int lane = threadIdx.x & 31;
    const int wave = threadIdx.x >> 5;
    const int row0 = blockIdx.x * 128 + wave * 16;
    const int hi   = lane >> 4;           // lane group 0..15 vs 16..31

    int arow = row0 + (lane & 15);
    if (arow >= nrows) arow = nrows - 1;  // clamp; stores handle bounds
    const __bf16* __restrict__ Ap = A + (size_t)arow * 128;

    // Preload all 4 K-chunks of the A fragment.
    // 16-bit A 16x32 layout: lanes 0-15 hold K = [K0..K0+7, K0+16..K0+23],
    //                        lanes 16-31 hold K = [K0+8..K0+15, K0+24..K0+31]
    v16bf afrag[4];
#pragma unroll
    for (int kc = 0; kc < 4; ++kc) {
        const int koff = kc * 32 + hi * 8;
        v8bf lo = *(const v8bf*)(Ap + koff);
        v8bf hh = *(const v8bf*)(Ap + koff + 16);
#pragma unroll
        for (int i = 0; i < 8; ++i) { afrag[kc][i] = lo[i]; afrag[kc][i + 8] = hh[i]; }
    }

    // Wave-uniform bounds check, forced into an SGPR so the store-path select
    // is a scalar branch (s_cmp/s_cbranch) instead of exec-mask juggling.
    const int fullstore =
        __builtin_amdgcn_readfirstlane((row0 + 16 <= nrows) ? 1 : 0);

    for (int ct = 0; ct < NOUT / 16; ++ct) {
        const int col = ct * 16 + (lane & 15);
        // 16-bit B 32x16 layout: lane (= column) holds K matching A:
        // lanes 0-15 -> K0..K0+7, K0+16..K0+23; lanes 16-31 -> K0+8..K0+15, K0+24..K0+31
        const __bf16* sBp = sB + col * SBS + hi * 8;

        v16bf bfrag[4];
#pragma unroll
        for (int kc = 0; kc < 4; ++kc) {
            v8bf blo = *(const v8bf*)(sBp + kc * 32);
            v8bf bhi = *(const v8bf*)(sBp + kc * 32 + 16);
#pragma unroll
            for (int i = 0; i < 8; ++i) { bfrag[kc][i] = blo[i]; bfrag[kc][i + 8] = bhi[i]; }
        }

        v8f acc = {};
#pragma unroll
        for (int kc = 0; kc < 4; ++kc) {
            acc = __builtin_amdgcn_wmma_f32_16x16x32_bf16(
                /*neg_a=*/false, afrag[kc], /*neg_b=*/false, bfrag[kc],
                /*c_mod=*/(short)0, acc, /*reuse_a=*/false, /*reuse_b=*/false);
        }

        // f32 C/D 16x16 layout: lane l holds column (l&15); VGPR v is row
        // v + (l>=16 ? 8 : 0).
        float* __restrict__ Cp = C + (size_t)(row0 + hi * 8) * NOUT + col;
        if (fullstore) {
#pragma unroll
            for (int v = 0; v < 8; ++v) Cp[(size_t)v * NOUT] = acc[v];
        } else {
            const int mbase = row0 + hi * 8;
#pragma unroll
            for (int v = 0; v < 8; ++v)
                if (mbase + v < nrows) Cp[(size_t)v * NOUT] = acc[v];
        }
    }
}

// ---------------------------------------------------------------------------
// Aggregation: agg[i] = dinv[i]^2 * t[i]  (self-loop term)
// ---------------------------------------------------------------------------
__global__ void k_selfloop(const float* __restrict__ t, const float* __restrict__ dinv,
                           float* __restrict__ agg, int n, int NOUT) {
    int id = blockIdx.x * blockDim.x + threadIdx.x;
    if (id < n * NOUT) {
        int i = id / NOUT;
        float di = dinv[i];
        agg[id] = t[id] * di * di;
    }
}

// agg[dst] += dinv[src]*dinv[dst] * t[src], one thread per (edge, 4 channels)
__global__ void k_edge_scatter(const int* __restrict__ ei, const float* __restrict__ t,
                               const float* __restrict__ dinv, float* __restrict__ agg,
                               int E, int NOUT) {
    const int cg = NOUT >> 2;  // channel groups of 4
    int gid = blockIdx.x * blockDim.x + threadIdx.x;
    if (gid >= E * cg) return;
    int e = gid / cg;
    int c = (gid - e * cg) << 2;
    int s = ei[e];
    int d = ei[E + e];
    float nrm = dinv[s] * dinv[d];
    const float4 v = *(const float4*)(t + (size_t)s * NOUT + c);
    float* __restrict__ o = agg + (size_t)d * NOUT + c;
    atomicAdd(o + 0, v.x * nrm);
    atomicAdd(o + 1, v.y * nrm);
    atomicAdd(o + 2, v.z * nrm);
    atomicAdd(o + 3, v.w * nrm);
}

// out_bf = bf16(relu(agg + b))  -> activation for the next GEMM
__global__ void k_bias_relu_bf(const float* __restrict__ agg, const float* __restrict__ b,
                               __bf16* __restrict__ out, int n, int NOUT) {
    int id = blockIdx.x * blockDim.x + threadIdx.x;
    if (id < n * NOUT) {
        float v = agg[id] + b[id % NOUT];
        out[id] = f2bf(v > 0.0f ? v : 0.0f);
    }
}

// final: h = agg + b3 (no relu), then row-wise L2 normalize (EMB = 64).
// One wave32 per node; each lane owns 2 channels (coalesced float2), sum of
// squares reduced across the wave with shfl_xor.
__global__ void k_final_norm(const float* __restrict__ agg, const float* __restrict__ b,
                             float* __restrict__ out, int n) {
    const int gid  = blockIdx.x * blockDim.x + threadIdx.x;
    const int node = gid >> 5;           // one wave per node
    const int lane = threadIdx.x & 31;
    if (node >= n) return;               // wave-uniform exit

    const int c = lane * 2;
    const float2 v = *(const float2*)(agg + (size_t)node * EMB + c);
    const float h0 = v.x + b[c];
    const float h1 = v.y + b[c + 1];

    float ss = h0 * h0 + h1 * h1;
#pragma unroll
    for (int m = 16; m > 0; m >>= 1) ss += __shfl_xor(ss, m, 32);

    float nrm = sqrtf(ss);
    nrm = nrm > 1e-12f ? nrm : 1e-12f;
    const float inv = 1.0f / nrm;

    float2 o;
    o.x = h0 * inv;
    o.y = h1 * inv;
    *(float2*)(out + (size_t)node * EMB + c) = o;
}

// ---------------------------------------------------------------------------
// Orchestration
// ---------------------------------------------------------------------------
static inline size_t alignup(size_t x) { return (x + 255) & ~(size_t)255; }

extern "C" void kernel_launch(void* const* d_in, const int* in_sizes, int n_in,
                              void* d_out, int out_size, void* d_ws, size_t ws_size,
                              hipStream_t stream) {
    const float* x   = (const float*)d_in[0];
    const int*   ei  = (const int*)  d_in[1];
    const float* W1  = (const float*)d_in[2];
    const float* b1  = (const float*)d_in[3];
    const float* W2  = (const float*)d_in[4];
    const float* b2  = (const float*)d_in[5];
    const float* W3  = (const float*)d_in[6];
    const float* b3  = (const float*)d_in[7];

    const int N = in_sizes[0] / IN_CH;   // 100000
    const int E = in_sizes[1] / 2;       // 640000

    // workspace carve-up
    char* ws = (char*)d_ws;
    size_t off = 0;
    float*  dinv = (float*)(ws + off);  off += alignup((size_t)N * 4);
    __bf16* actA = (__bf16*)(ws + off); off += alignup((size_t)N * HID * 2);
    __bf16* actB = (__bf16*)(ws + off); off += alignup((size_t)N * HID * 2);
    float*  tbuf = (float*)(ws + off);  off += alignup((size_t)N * HID * 4);
    float*  agg  = (float*)(ws + off);  off += alignup((size_t)N * HID * 4);
    __bf16* W1t  = (__bf16*)(ws + off); off += alignup((size_t)IN_CH * HID * 2);
    __bf16* W2t  = (__bf16*)(ws + off); off += alignup((size_t)HID * HID * 2);
    __bf16* W3t  = (__bf16*)(ws + off); off += alignup((size_t)HID * EMB * 2);
    (void)ws_size;

    const int T = 256;
    auto blks = [](long long n, int t) { return (int)((n + t - 1) / t); };

    // ---- gcn_norm: degrees (self-loops included), dinv = rsqrt(deg) ----
    k_init_one<<<blks(N, T), T, 0, stream>>>(dinv, N);
    k_degree  <<<blks(E, T), T, 0, stream>>>(ei, dinv, E);
    k_rsqrt   <<<blks(N, T), T, 0, stream>>>(dinv, N);

    // ---- input + weight conversion ----
    k_f2bf<<<blks((long long)N * IN_CH, T), T, 0, stream>>>(x, actA, N * IN_CH);
    k_transpose_bf<<<blks(IN_CH * HID, T), T, 0, stream>>>(W1, W1t, IN_CH, HID);
    k_transpose_bf<<<blks(HID * HID, T), T, 0, stream>>>(W2, W2t, HID, HID);
    k_transpose_bf<<<blks(HID * EMB, T), T, 0, stream>>>(W3, W3t, HID, EMB);

    const int gemm_grid = blks(N, 128);

    // ---- layer 1: t = actA@W1; agg = scatter(norm*t); actB = bf16(relu(agg+b1)) ----
    k_gemm_bf16_wmma<HID><<<gemm_grid, 256, 0, stream>>>(actA, W1t, tbuf, N);
    k_selfloop<<<blks((long long)N * HID, T), T, 0, stream>>>(tbuf, dinv, agg, N, HID);
    k_edge_scatter<<<blks((long long)E * (HID / 4), T), T, 0, stream>>>(ei, tbuf, dinv, agg, E, HID);
    k_bias_relu_bf<<<blks((long long)N * HID, T), T, 0, stream>>>(agg, b1, actB, N, HID);

    // ---- layer 2 ----
    k_gemm_bf16_wmma<HID><<<gemm_grid, 256, 0, stream>>>(actB, W2t, tbuf, N);
    k_selfloop<<<blks((long long)N * HID, T), T, 0, stream>>>(tbuf, dinv, agg, N, HID);
    k_edge_scatter<<<blks((long long)E * (HID / 4), T), T, 0, stream>>>(ei, tbuf, dinv, agg, E, HID);
    k_bias_relu_bf<<<blks((long long)N * HID, T), T, 0, stream>>>(agg, b2, actA, N, HID);

    // ---- layer 3 (128 -> 64), then L2 row-normalize ----
    k_gemm_bf16_wmma<EMB><<<gemm_grid, 256, 0, stream>>>(actA, W3t, tbuf, N);
    k_selfloop<<<blks((long long)N * EMB, T), T, 0, stream>>>(tbuf, dinv, agg, N, EMB);
    k_edge_scatter<<<blks((long long)E * (EMB / 4), T), T, 0, stream>>>(ei, tbuf, dinv, agg, E, EMB);
    k_final_norm<<<blks((long long)N * 32, T), T, 0, stream>>>(agg, b3, (float*)d_out, N);
}
